// AE_62010737820083
// MI455X (gfx1250) — matrix operand order repaired and checked by reading, not compiled, not run on hardware
//
#include <hip/hip_runtime.h>
#include <hip/hip_bf16.h>
#include <math.h>

// VQ-VAE forward for MI455X (gfx1250, wave32).
// 64-ch convs / transposed-convs run as implicit GEMM on v_wmma_f32_16x16x32_f16
// (f16 data, f32 accumulate). Intermediates kept in f16 to halve HBM traffic
// (network is memory-bound: ~33 GFLOP vs ~0.5+ GB of activation traffic).
// M-tile = 32 rows: each wave holds two f32 accumulators and issues 2 WMMA per
// B-fragment load; all 128 threads stage the 2KB LDS A-tile (no exec branch).

typedef __attribute__((ext_vector_type(16))) _Float16 v16h;
typedef __attribute__((ext_vector_type(8)))  _Float16 v8h;
typedef __attribute__((ext_vector_type(8)))  float    v8f;

union AFrag { v16h v; v8h h[2]; };

// ---------------------------------------------------------------------------
// Weight shuffle: f32 HWIO [3,3,64,64] -> f16 WMMA-B fragments.
// dst[(((t*2+h)*4+n)*32+lane)*16+v] = W[t, h*32 + ((v&8)?16:0)+((lane>=16)?8:0)+(v&7), n*16+(lane&15)]
// so each lane's 16 f16 B-fragment values are one contiguous 32B load.
// ---------------------------------------------------------------------------
__global__ __launch_bounds__(256)
void k_wconv(const float* __restrict__ src, _Float16* __restrict__ dst) {
    int e = blockIdx.x * 256 + threadIdx.x;
    if (e >= 9 * 2 * 4 * 32 * 16) return;
    int v    = e & 15;
    int lane = (e >> 4) & 31;
    int n    = (e >> 9) & 3;
    int h    = (e >> 11) & 1;
    int t    = e >> 12;
    int K  = ((v >> 3) << 4) + ((lane >> 4) << 3) + (v & 7);      // lanes 16-31 hold K={8..15,24..31}
    int ci = h * 32 + K;
    int co = n * 16 + (lane & 15);
    dst[e] = (_Float16)src[(t * 64 + ci) * 64 + co];
}

// ---------------------------------------------------------------------------
// Strided conv, 3x3 s2 SAME (pad lo=0, hi=1), 64->64, ReLU. Implicit GEMM:
// M = B*Hout*Wout, N = 64, K = 576. Block = 4 waves; wave w owns couts
// [16w,16w+16). 32-row A chunk staged in LDS per (tap, ci-half); each wave
// runs two 16x16x32 WMMAs per chunk (rows 0-15 and 16-31).
// ---------------------------------------------------------------------------
__global__ __launch_bounds__(128)
void k_conv_s2_wmma(const _Float16* __restrict__ in, const _Float16* __restrict__ wb,
                    const float* __restrict__ bias, _Float16* __restrict__ out,
                    int Hin, int Win, int Hout, int Wout) {
    __shared__ __align__(16) _Float16 At[32 * 32];
    const int tid   = threadIdx.x;
    const int lane  = tid & 31;
    const int wave  = tid >> 5;
    const int mbase = blockIdx.x * 32;

    // Staging coordinates for this thread's row (K-invariant).
    const int sr  = tid >> 2;           // row 0..31
    const int sc8 = (tid & 3) * 8;      // f16 column offset 0/8/16/24
    const int sm  = mbase + sr;
    const int sx  = sm % Wout;
    const int smy = sm / Wout;
    const int sy  = smy % Hout;
    const int sb  = smy / Hout;
    const _Float16* sbase = in + ((size_t)sb * Hin) * Win * 64;

    // A-fragment LDS addresses (ISA 16-bit A 16x32 layout):
    // lanes 0-15: row=lane, K={0..7,16..23}; lanes 16-31: row=lane-16, K={8..15,24..31}
    const int arow = lane & 15;
    const int s    = (lane >> 4) * 8;

    v8f acc0 = {}, acc1 = {};
    for (int t = 0; t < 9; ++t) {
        const int dy = t / 3, dx = t % 3;
        const int iy = 2 * sy + dy, ix = 2 * sx + dx;
        const bool ok = (iy < Hin) & (ix < Win);
        const _Float16* srcp = sbase + ((size_t)iy * Win + ix) * 64 + sc8;
        for (int h = 0; h < 2; ++h) {
            __syncthreads();
            v8h val = {};
            if (ok) val = *(const v8h*)(srcp + h * 32);
            *(v8h*)(At + sr * 32 + sc8) = val;
            __syncthreads();
            AFrag a0, a1;
            a0.h[0] = *(const v8h*)(At + arow * 32 + s);
            a0.h[1] = *(const v8h*)(At + arow * 32 + 16 + s);
            a1.h[0] = *(const v8h*)(At + (16 + arow) * 32 + s);
            a1.h[1] = *(const v8h*)(At + (16 + arow) * 32 + 16 + s);
            const v16h bfrag = *(const v16h*)(wb + ((((t * 2 + h) * 4 + wave) * 32 + lane) << 4));
            acc0 = __builtin_amdgcn_wmma_f32_16x16x32_f16(false, a0.v, false, bfrag,
                                                          (short)0, acc0, false, false);
            acc1 = __builtin_amdgcn_wmma_f32_16x16x32_f16(false, a1.v, false, bfrag,
                                                          (short)0, acc1, false, false);
        }
    }
    const int   co = wave * 16 + (lane & 15);
    const float bv = bias[co];
    const int   mr = (lane >> 4) * 8;   // C/D layout: VGPR i -> M = i (+8 for hi lanes)
    for (int i = 0; i < 8; ++i) {
        float v0 = fmaxf(acc0[i] + bv, 0.0f);
        float v1 = fmaxf(acc1[i] + bv, 0.0f);
        out[(size_t)(mbase + mr + i) * 64 + co]      = (_Float16)v0;
        out[(size_t)(mbase + 16 + mr + i) * 64 + co] = (_Float16)v1;
    }
}

// ---------------------------------------------------------------------------
// Transposed conv, 3x3 s2 SAME (lhs-dilated conv, pads (2,1), no kernel flip),
// 64->64, ReLU. blockIdx.y = output parity class (py,px); within a class the
// tap set is uniform so it is again an implicit GEMM with K = taps*64.
//   py==0: dy in {0 (iy=y'-1), 2 (iy=y')};  py==1: dy==1 (iy=y')
// ---------------------------------------------------------------------------
__global__ __launch_bounds__(128)
void k_convT_s2_wmma(const _Float16* __restrict__ in, const _Float16* __restrict__ wb,
                     const float* __restrict__ bias, _Float16* __restrict__ out,
                     int Hin, int Win) {
    __shared__ __align__(16) _Float16 At[32 * 32];
    const int tid   = threadIdx.x;
    const int lane  = tid & 31;
    const int wave  = tid >> 5;
    const int mbase = blockIdx.x * 32;
    const int pc = blockIdx.y;
    const int py = pc >> 1, px = pc & 1;
    const int nty = py ? 1 : 2, ntx = px ? 1 : 2;
    const int Hout = Hin * 2, Wout = Win * 2;

    const int sr  = tid >> 2;
    const int sc8 = (tid & 3) * 8;
    const int sm  = mbase + sr;
    const int sxq = sm % Win;
    const int smy = sm / Win;
    const int syq = smy % Hin;
    const int sb  = smy / Hin;
    const _Float16* sbase = in + ((size_t)sb * Hin) * Win * 64;

    const int arow = lane & 15;
    const int s    = (lane >> 4) * 8;

    v8f acc0 = {}, acc1 = {};
    for (int ty = 0; ty < nty; ++ty) {
        const int dy = py ? 1 : ty * 2;
        const int yo = py ? 0 : ty - 1;
        for (int tx = 0; tx < ntx; ++tx) {
            const int dx = px ? 1 : tx * 2;
            const int xo = px ? 0 : tx - 1;
            const int t  = dy * 3 + dx;
            const int iy = syq + yo, ix = sxq + xo;
            const bool ok = (iy >= 0) & (ix >= 0);
            const _Float16* srcp = sbase + ((size_t)iy * Win + ix) * 64 + sc8;
            for (int h = 0; h < 2; ++h) {
                __syncthreads();
                v8h val = {};
                if (ok) val = *(const v8h*)(srcp + h * 32);
                *(v8h*)(At + sr * 32 + sc8) = val;
                __syncthreads();
                AFrag a0, a1;
                a0.h[0] = *(const v8h*)(At + arow * 32 + s);
                a0.h[1] = *(const v8h*)(At + arow * 32 + 16 + s);
                a1.h[0] = *(const v8h*)(At + (16 + arow) * 32 + s);
                a1.h[1] = *(const v8h*)(At + (16 + arow) * 32 + 16 + s);
                const v16h bfrag = *(const v16h*)(wb + ((((t * 2 + h) * 4 + wave) * 32 + lane) << 4));
                acc0 = __builtin_amdgcn_wmma_f32_16x16x32_f16(false, a0.v, false, bfrag,
                                                              (short)0, acc0, false, false);
                acc1 = __builtin_amdgcn_wmma_f32_16x16x32_f16(false, a1.v, false, bfrag,
                                                              (short)0, acc1, false, false);
            }
        }
    }
    const int   co = wave * 16 + (lane & 15);
    const float bv = bias[co];
    const int   mr = (lane >> 4) * 8;
    for (int i = 0; i < 8; ++i) {
        for (int tile = 0; tile < 2; ++tile) {
            float v = fmaxf((tile ? acc1[i] : acc0[i]) + bv, 0.0f);
            const int m  = mbase + tile * 16 + mr + i;
            const int xq = m % Win;
            const int my = m / Win;
            const int yq = my % Hin;
            const int b  = my / Hin;
            const int oy = 2 * yq + py, ox = 2 * xq + px;
            out[(((size_t)(b * Hout + oy)) * Wout + ox) * 64 + co] = (_Float16)v;
        }
    }
}

// ---- enc0: 3x3 s2, 1->64, ReLU; x f32 [16,256,256,1] -> f16 [16,128,128,64]
__global__ __launch_bounds__(256)
void k_enc0(const float* __restrict__ x, const float* __restrict__ w,
            const float* __restrict__ bias, _Float16* __restrict__ out) {
    const int idx = blockIdx.x * 256 + threadIdx.x;   // 16*128*128*64
    const int co = idx & 63;
    const int m  = idx >> 6;
    const int xo = m & 127;
    const int yo = (m >> 7) & 127;
    const int b  = m >> 14;
    float acc = bias[co];
    for (int dy = 0; dy < 3; ++dy) {
        const int iy = 2 * yo + dy;
        if (iy >= 256) continue;
        for (int dx = 0; dx < 3; ++dx) {
            const int ix = 2 * xo + dx;
            if (ix >= 256) continue;
            acc += x[((size_t)(b * 256 + iy)) * 256 + ix] * w[(dy * 3 + dx) * 64 + co];
        }
    }
    out[idx] = (_Float16)fmaxf(acc, 0.0f);
}

// ---- latent: 3x3 s1 SAME, 64->8, ReLU; f16 [16,4,4,64] -> f32 [16,4,4,8]
__global__ __launch_bounds__(256)
void k_lat(const _Float16* __restrict__ in, const float* __restrict__ w,
           const float* __restrict__ bias, float* __restrict__ out) {
    const int idx = blockIdx.x * 256 + threadIdx.x;
    if (idx >= 16 * 4 * 4 * 8) return;
    const int co = idx & 7;
    const int m  = idx >> 3;
    const int xo = m & 3;
    const int yo = (m >> 2) & 3;
    const int b  = m >> 4;
    float acc = bias[co];
    for (int dy = 0; dy < 3; ++dy) {
        const int iy = yo + dy - 1;
        if (iy < 0 || iy >= 4) continue;
        for (int dx = 0; dx < 3; ++dx) {
            const int ix = xo + dx - 1;
            if (ix < 0 || ix >= 4) continue;
            const _Float16* p  = in + ((b * 4 + iy) * 4 + ix) * 64;
            const float*    wp = w + ((dy * 3 + dx) * 64) * 8 + co;
            for (int ci = 0; ci < 64; ++ci)
                acc += (float)p[ci] * wp[ci * 8];
        }
    }
    out[idx] = fmaxf(acc, 0.0f);
}

// ---- VQ: nearest codebook entry (first-min, matches jnp.argmin), q = emb[idx]
__global__ __launch_bounds__(64)
void k_vq(const float* __restrict__ lat, const float* __restrict__ emb,
          float* __restrict__ q) {
    const int pos = blockIdx.x * 64 + threadIdx.x;
    if (pos >= 256) return;
    float f[8];
    for (int j = 0; j < 8; ++j) f[j] = lat[pos * 8 + j];
    float best = 3.4e38f; int bi = 0;
    for (int k = 0; k < 64; ++k) {
        float d = 0.f;
        for (int j = 0; j < 8; ++j) { float t = f[j] - emb[k * 8 + j]; d += t * t; }
        if (d < best) { best = d; bi = k; }
    }
    for (int j = 0; j < 8; ++j) q[pos * 8 + j] = emb[bi * 8 + j];
}

// ---- dec0: convT 3x3 s2, 8->64, ReLU; q f32 [16,4,4,8] -> f16 [16,8,8,64]
__global__ __launch_bounds__(256)
void k_dec0(const float* __restrict__ q, const float* __restrict__ w,
            const float* __restrict__ bias, _Float16* __restrict__ out) {
    const int idx = blockIdx.x * 256 + threadIdx.x;   // 16*8*8*64
    const int co = idx & 63;
    const int m  = idx >> 6;
    const int ox = m & 7;
    const int oy = (m >> 3) & 7;
    const int b  = m >> 6;
    float acc = bias[co];
    for (int dy = 0; dy < 3; ++dy) {
        const int p = oy + dy - 2;
        if (p < 0 || (p & 1)) continue;
        const int iy = p >> 1;
        if (iy >= 4) continue;
        for (int dx = 0; dx < 3; ++dx) {
            const int px = ox + dx - 2;
            if (px < 0 || (px & 1)) continue;
            const int ix = px >> 1;
            if (ix >= 4) continue;
            const float* ip = q + ((b * 4 + iy) * 4 + ix) * 8;
            const float* wp = w + ((dy * 3 + dx) * 8) * 64 + co;
            for (int ci = 0; ci < 8; ++ci)
                acc += ip[ci] * wp[ci * 64];
        }
    }
    out[idx] = (_Float16)fmaxf(acc, 0.0f);
}

// ---- output: convT 3x3 s1 SAME (== conv pad 1), 64->1, sigmoid -> f32 d_out
__global__ __launch_bounds__(256)
void k_outc(const _Float16* __restrict__ in, const float* __restrict__ w,
            const float* __restrict__ bias, float* __restrict__ out) {
    const int idx = blockIdx.x * 256 + threadIdx.x;   // 16*256*256
    const int ox = idx & 255;
    const int oy = (idx >> 8) & 255;
    const int b  = idx >> 16;
    float acc = bias[0];
    for (int dy = 0; dy < 3; ++dy) {
        const int iy = oy + dy - 1;
        if (iy < 0 || iy >= 256) continue;
        for (int dx = 0; dx < 3; ++dx) {
            const int ix = ox + dx - 1;
            if (ix < 0 || ix >= 256) continue;
            const _Float16* ip = in + (((size_t)(b * 256 + iy)) * 256 + ix) * 64;
            const float*    wp = w + (dy * 3 + dx) * 64;
            float s = 0.f;
            for (int ci = 0; ci < 64; ++ci) s += (float)ip[ci] * wp[ci];
            acc += s;
        }
    }
    out[idx] = 1.0f / (1.0f + expf(-acc));
}

extern "C" void kernel_launch(void* const* d_in, const int* in_sizes, int n_in,
                              void* d_out, int out_size, void* d_ws, size_t ws_size,
                              hipStream_t stream) {
    (void)in_sizes; (void)n_in; (void)out_size; (void)ws_size;
    const float* x = (const float*)d_in[0];
    const float* enc_k[6]; const float* enc_b[6];
    for (int n = 0; n < 6; ++n) { enc_k[n] = (const float*)d_in[1 + 2 * n];
                                  enc_b[n] = (const float*)d_in[2 + 2 * n]; }
    const float* lat_k = (const float*)d_in[13];
    const float* lat_b = (const float*)d_in[14];
    const float* dec_k[6]; const float* dec_b[6];
    for (int n = 0; n < 6; ++n) { dec_k[n] = (const float*)d_in[15 + 2 * n];
                                  dec_b[n] = (const float*)d_in[16 + 2 * n]; }
    const float* out_k = (const float*)d_in[27];
    const float* out_b = (const float*)d_in[28];
    const float* emb   = (const float*)d_in[29];

    // Workspace: [10 x f16 WMMA weights 73728B][bufB 32MiB][bufA 128MiB]
    char* ws = (char*)d_ws;
    const size_t WELEMS = 36864;
    _Float16* wenc[5]; _Float16* wdec[5];
    for (int i = 0; i < 5; ++i) wenc[i] = (_Float16*)ws + (size_t)i * WELEMS;
    for (int i = 0; i < 5; ++i) wdec[i] = (_Float16*)ws + (size_t)(5 + i) * WELEMS;
    char* base = ws + 10 * WELEMS * 2;                    // 737280 B
    _Float16* bufB = (_Float16*)base;                     // 33,554,432 B
    _Float16* bufA = (_Float16*)(base + 33554432ull);     // 134,217,728 B

    for (int i = 0; i < 5; ++i) k_wconv<<<144, 256, 0, stream>>>(enc_k[i + 1], wenc[i]);
    for (int i = 0; i < 5; ++i) k_wconv<<<144, 256, 0, stream>>>(dec_k[i + 1], wdec[i]);

    // Encoder (ping-pong B <-> A); grid.x = M_total/32
    k_enc0<<<65536, 256, 0, stream>>>(x, enc_k[0], enc_b[0], bufB);
    k_conv_s2_wmma<<<2048, 128, 0, stream>>>(bufB, wenc[0], enc_b[1], bufA, 128, 128, 64, 64);
    k_conv_s2_wmma<<<512,  128, 0, stream>>>(bufA, wenc[1], enc_b[2], bufB, 64, 64, 32, 32);
    k_conv_s2_wmma<<<128,  128, 0, stream>>>(bufB, wenc[2], enc_b[3], bufA, 32, 32, 16, 16);
    k_conv_s2_wmma<<<32,   128, 0, stream>>>(bufA, wenc[3], enc_b[4], bufB, 16, 16, 8, 8);
    k_conv_s2_wmma<<<8,    128, 0, stream>>>(bufB, wenc[4], enc_b[5], bufA, 8, 8, 4, 4);
    k_lat<<<8, 256, 0, stream>>>(bufA, lat_k, lat_b, (float*)bufB);
    k_vq<<<4, 64, 0, stream>>>((const float*)bufB, emb, (float*)bufA);

    // Decoder; grid = (class_M/32, 4 parity classes)
    k_dec0<<<256, 256, 0, stream>>>((const float*)bufA, dec_k[0], dec_b[0], bufB);
    k_convT_s2_wmma<<<dim3(32,   4), 128, 0, stream>>>(bufB, wdec[0], dec_b[1], bufA, 8, 8);
    k_convT_s2_wmma<<<dim3(128,  4), 128, 0, stream>>>(bufA, wdec[1], dec_b[2], bufB, 16, 16);
    k_convT_s2_wmma<<<dim3(512,  4), 128, 0, stream>>>(bufB, wdec[2], dec_b[3], bufA, 32, 32);
    k_convT_s2_wmma<<<dim3(2048, 4), 128, 0, stream>>>(bufA, wdec[3], dec_b[4], bufB, 64, 64);
    k_convT_s2_wmma<<<dim3(8192, 4), 128, 0, stream>>>(bufB, wdec[4], dec_b[5], bufA, 128, 128);
    k_outc<<<4096, 256, 0, stream>>>(bufA, out_k, out_b, (float*)d_out);
}
